// MultiHeadSelfAttention_65867618452066
// MI455X (gfx1250) — matrix operand order repaired and checked by reading, compiled, not run on hardware
//
#include <hip/hip_runtime.h>
#include <hip/hip_bf16.h>

// ---------------------------------------------------------------------------
// MHA forward for B=2, T=2048, D_MODEL=1024, H=16, dh=64 on gfx1250 (CDNA5).
// bf16 WMMA (v_wmma_f32_16x16x32_bf16) with f32 accumulation everywhere.
// Flash attention computes S^T so that (a) softmax rows are lane-resident and
// (b) the P^T C-layout coincides exactly with the WMMA A-layout. K-chunk
// staging uses gfx1250 GLOBAL_LOAD_ASYNC_TO_LDS (ASYNCcnt-tracked DMA).
// ---------------------------------------------------------------------------

typedef __bf16 bf16_t;
typedef __attribute__((ext_vector_type(16))) bf16_t v16bf;
typedef __attribute__((ext_vector_type(8)))  bf16_t v8bf;
typedef __attribute__((ext_vector_type(8)))  float  v8f;

#define T_SEQ   2048
#define N_HEADS 16
#define D_HEAD  64
#define C_MODEL 1024
#define QKV_C   3072   // 3 * C_MODEL, row stride of qkv activation
#define BATCH   2

static __device__ __forceinline__ v8f vzero8() {
  v8f z = {0.f, 0.f, 0.f, 0.f, 0.f, 0.f, 0.f, 0.f};
  return z;
}

static __device__ __forceinline__ v16bf cat16(v8bf lo, v8bf hi) {
  return __builtin_shufflevector(lo, hi, 0,1,2,3,4,5,6,7,8,9,10,11,12,13,14,15);
}

static __device__ __forceinline__ v8f wmma_bf16(v16bf a, v16bf b, v8f c) {
  // D = A(16x32 bf16) * B(32x16 bf16) + C(16x16 f32)
  return __builtin_amdgcn_wmma_f32_16x16x32_bf16(false, a, false, b, (short)0, c,
                                                 false, false);
}

// A-fragment (16x32, interleaved CDNA5 A layout):
//  lanes 0-15 : row = l,    K = {0..7, 16..23}
//  lanes 16-31: row = l-16, K = {8..15, 24..31}
static __device__ __forceinline__ v16bf load_afrag(const bf16_t* rowptr, int sel) {
  v8bf lo = *(const v8bf*)(rowptr + sel * 8);
  v8bf hi = *(const v8bf*)(rowptr + 16 + sel * 8);
  return cat16(lo, hi);
}

// B-fragment (32x16, interleaved layout matching A):
//  lanes 0-15 hold column n = lane, K = {0..7, 16..23} ; lanes 16-31 hold K = {8..15, 24..31}.
//  `colptr` must already include the +sel*8 K offset.
static __device__ __forceinline__ v16bf load_bfrag(const bf16_t* colptr) {
  v8bf lo = *(const v8bf*)(colptr);
  v8bf hi = *(const v8bf*)(colptr + 16);
  return cat16(lo, hi);
}

// gfx1250 async global->LDS copy (16B per lane), tracked by ASYNCcnt.
// LDS offset = low 32 bits of the generic shared-aperture address.
static __device__ __forceinline__ void async_g2l_b128(void* lds_ptr,
                                                      const void* gptr) {
  unsigned lds_off = (unsigned)(unsigned long long)lds_ptr;
  asm volatile("global_load_async_to_lds_b128 %0, %1, off"
               :
               : "v"(lds_off), "v"(gptr)
               : "memory");
}
static __device__ __forceinline__ void wait_asynccnt0() {
  asm volatile("s_wait_asynccnt 0x0" ::: "memory");
}

// ---------------------------------------------------------------------------
// f32 -> bf16 bulk conversion
// ---------------------------------------------------------------------------
__global__ void cvt_f32_to_bf16(const float* __restrict__ in,
                                bf16_t* __restrict__ out, int n) {
  int i = (blockIdx.x * blockDim.x + threadIdx.x) * 4;
  if (i + 3 < n) {
    float4 v = *(const float4*)(in + i);
    out[i + 0] = (bf16_t)v.x;
    out[i + 1] = (bf16_t)v.y;
    out[i + 2] = (bf16_t)v.z;
    out[i + 3] = (bf16_t)v.w;
  } else {
    for (; i < n; ++i) out[i] = (bf16_t)in[i];
  }
}

// ---------------------------------------------------------------------------
// GEMM: C[M,N] = A[M,K] * W[N,K]^T + bias[N]
// One wave computes a 64x64 tile (4x4 WMMA accumulators), K stepped by 32.
// 8 waves / block. All dims are multiples of 64 here. Working set is
// L2-resident (192 MB), so direct global fragment loads suffice.
// ---------------------------------------------------------------------------
template <bool OUT_BF16>
__global__ __launch_bounds__(256)
void gemm_wmma(const bf16_t* __restrict__ A, const bf16_t* __restrict__ W,
               const float* __restrict__ bias, bf16_t* __restrict__ Cb,
               float* __restrict__ Cf, int M, int N, int K) {
  const int lane = threadIdx.x & 31;
  const int wv   = __builtin_amdgcn_readfirstlane(threadIdx.x >> 5);
  const int nt   = N >> 6;
  const int mt   = M >> 6;
  const int tile = blockIdx.x * 8 + wv;     // scalar -> scalar branch below
  if (tile >= mt * nt) return;              // wave-uniform exit

  const int m0  = (tile / nt) << 6;
  const int n0  = (tile % nt) << 6;
  const int sel = (lane >> 4) & 1;
  const int l15 = lane & 15;

  v8f acc[4][4];
#pragma unroll
  for (int i = 0; i < 4; ++i)
#pragma unroll
    for (int j = 0; j < 4; ++j) acc[i][j] = vzero8();

  const bf16_t* arow[4];
  const bf16_t* brow[4];
#pragma unroll
  for (int ci = 0; ci < 4; ++ci)
    arow[ci] = A + (size_t)(m0 + 16 * ci + l15) * K;
#pragma unroll
  for (int cj = 0; cj < 4; ++cj)
    brow[cj] = W + (size_t)(n0 + 16 * cj + l15) * K + sel * 8;

#pragma unroll 2
  for (int k0 = 0; k0 < K; k0 += 32) {
    v16bf af[4], bf[4];
#pragma unroll
    for (int ci = 0; ci < 4; ++ci) af[ci] = load_afrag(arow[ci] + k0, sel);
#pragma unroll
    for (int cj = 0; cj < 4; ++cj) bf[cj] = load_bfrag(brow[cj] + k0);
#pragma unroll
    for (int ci = 0; ci < 4; ++ci)
#pragma unroll
      for (int cj = 0; cj < 4; ++cj)
        acc[ci][cj] = wmma_bf16(af[ci], bf[cj], acc[ci][cj]);
  }

  // C layout: VGPR v holds (row = v + 8*sel, col = lane&15) of each 16x16 tile
#pragma unroll
  for (int cj = 0; cj < 4; ++cj) {
    const int col = n0 + 16 * cj + l15;
    const float bc = bias[col];
#pragma unroll
    for (int ci = 0; ci < 4; ++ci) {
#pragma unroll
      for (int v = 0; v < 8; ++v) {
        const int row = m0 + 16 * ci + v + 8 * sel;
        const float val = acc[ci][cj][v] + bc;
        if (OUT_BF16)
          Cb[(size_t)row * N + col] = (bf16_t)val;
        else
          Cf[(size_t)row * N + col] = val;
      }
    }
  }
}

// ---------------------------------------------------------------------------
// Flash attention: one block = (b, h, 64 q rows), 4 waves of 16 q rows each.
// K/V chunks of 32 rows are cooperatively staged in LDS (K via async DMA).
//
// S^T trick: computing S^T = K_chunk * Q^T gives a C-layout where lane l
// holds q-column (q0+l15) and VGPR v holds k-row (kb + 16j + v + 8*sel).
//  * softmax rows are lane-resident: 16 local elements + ONE xor-16 shuffle
//  * P^T's C-layout == P's A-layout (lane l: row q=l, k in {0-7,16-23} /
//    {8-15,24-31}) -> exp() results feed the P*V WMMA directly from registers
// qkv layout: [B*T, 3072] bf16, head h at columns h*192 + {q:0, k:64, v:128}.
// Output: attn_out [B*T, 1024] bf16, column = h*64 + d.
// ---------------------------------------------------------------------------
#define KC_STRIDE 72   // 32 x 64 K chunk, padded row stride (elements)
#define VT_STRIDE 40   // 64 x 32 transposed V chunk, padded row stride

__global__ __launch_bounds__(128)
void flash_attn(const bf16_t* __restrict__ qkv, bf16_t* __restrict__ attn_out) {
  __shared__ bf16_t Kc[32 * KC_STRIDE];
  __shared__ bf16_t Vt[64 * VT_STRIDE];

  const int tid  = threadIdx.x;
  const int lane = tid & 31;
  const int wv   = __builtin_amdgcn_readfirstlane(tid >> 5);
  const int sel  = (lane >> 4) & 1;
  const int l15  = lane & 15;

  const int bid = blockIdx.x;            // B * H * (T/64) = 1024 blocks
  const int qt  = bid & 31;              // q tile of 64 rows
  const int h   = (bid >> 5) & 15;
  const int b   = bid >> 9;

  const int q0 = qt * 64 + wv * 16;      // wave-uniform first q row
  const bf16_t* base = qkv + (size_t)b * T_SEQ * QKV_C + h * (3 * D_HEAD);

  // --- load Q as two B-fragments (lane = q column, d interleaved like A) ---
  const int qrow = q0 + l15;             // this lane's q row
  const bf16_t* qp = base + (size_t)qrow * QKV_C;  // q at column offset 0
  v16bf qB[2];
#pragma unroll
  for (int dc = 0; dc < 2; ++dc)
    qB[dc] = load_bfrag(qp + 32 * dc + sel * 8);

  v8f o[4];
#pragma unroll
  for (int j = 0; j < 4; ++j) o[j] = vzero8();

  // per-q-row stats, lane-resident (replicated across the two halves)
  float m_s = -__builtin_inff();
  float l_s = 0.f;

  const int nchunks = qt * 2 + 2;        // causal: cols up to q0_block + 63
  for (int ch = 0; ch < nchunks; ++ch) {
    const int kb = ch * 32;

    __syncthreads();  // previous chunk's LDS reads complete before overwrite
    {
      // cooperative stage: 128 threads x 32B of K (async DMA) and V each
      const int idx = tid * 16;
      const int kr  = idx >> 6;          // 0..31
      const int c0  = idx & 63;          // 0,16,32,48
      const bf16_t* kp = base + (size_t)(kb + kr) * QKV_C + D_HEAD + c0;
      async_g2l_b128(&Kc[kr * KC_STRIDE + c0], kp);
      async_g2l_b128(&Kc[kr * KC_STRIDE + c0 + 8], kp + 8);

      const bf16_t* vp = base + (size_t)(kb + kr) * QKV_C + 2 * D_HEAD + c0;
      v8bf va = *(const v8bf*)vp;
      v8bf vb = *(const v8bf*)(vp + 8);
#pragma unroll
      for (int i = 0; i < 8; ++i) {
        Vt[(c0 + i) * VT_STRIDE + kr]     = va[i];
        Vt[(c0 + 8 + i) * VT_STRIDE + kr] = vb[i];
      }
      wait_asynccnt0();                  // K DMA landed in LDS
    }
    __syncthreads();

    // wave-uniform skip: chunk entirely above this wave's diagonal
    if (kb > q0 + 15) continue;

    // --- S^T = K_chunk * Q^T  (element [j][v] = S[q=qrow][k=kb+16j+v+8*sel])
    v8f s[2];
#pragma unroll
    for (int j = 0; j < 2; ++j) {
      v8f a = vzero8();
#pragma unroll
      for (int dc = 0; dc < 2; ++dc) {
        const bf16_t* ap = &Kc[(16 * j + l15) * KC_STRIDE + 32 * dc];
        a = wmma_bf16(load_afrag(ap, sel), qB[dc], a);
      }
      s[j] = a;
    }

    // --- scale (+ causal mask only on the partial diagonal chunk) ---
    if (kb + 31 > q0) {
      const int klim = qrow - kb - 8 * sel;   // keep k-offset 16j+v <= klim
#pragma unroll
      for (int j = 0; j < 2; ++j)
#pragma unroll
        for (int v = 0; v < 8; ++v) {
          float x = s[j][v] * 0.125f;          // 1/sqrt(64)
          s[j][v] = (16 * j + v > klim) ? -__builtin_inff() : x;
        }
    } else {
#pragma unroll
      for (int j = 0; j < 2; ++j)
#pragma unroll
        for (int v = 0; v < 8; ++v) s[j][v] *= 0.125f;
    }

    // --- online softmax: row is lane-resident across the two halves ---
    float mc = fmaxf(s[0][0], s[1][0]);
#pragma unroll
    for (int v = 1; v < 8; ++v) mc = fmaxf(mc, fmaxf(s[0][v], s[1][v]));
    mc = fmaxf(mc, __shfl_xor(mc, 16, 32));    // combine halves
    const float mn = fmaxf(m_s, mc);
    const float alpha = __expf(m_s - mn);

    float ps = 0.f;
#pragma unroll
    for (int j = 0; j < 2; ++j)
#pragma unroll
      for (int v = 0; v < 8; ++v) {
        const float p = __expf(s[j][v] - mn);
        s[j][v] = p;
        ps += p;
      }
    ps += __shfl_xor(ps, 16, 32);              // combine halves
    l_s = l_s * alpha + ps;
    m_s = mn;

    // --- P^T C-layout == P A-layout: pack exp() results directly ---
    v16bf pf;
#pragma unroll
    for (int v = 0; v < 8; ++v) {
      pf[v]     = (bf16_t)s[0][v];
      pf[v + 8] = (bf16_t)s[1][v];
    }

    // --- rescale O by per-q alpha (O VGPR v is q-row v+8*sel -> shuffle) ---
    float av[8];
#pragma unroll
    for (int v = 0; v < 8; ++v) av[v] = __shfl(alpha, v + 8 * sel, 32);
#pragma unroll
    for (int cj = 0; cj < 4; ++cj)
#pragma unroll
      for (int v = 0; v < 8; ++v) o[cj][v] *= av[v];

    // --- O += P * V  (V: four 32x16 B-frags from transposed LDS tile) ---
#pragma unroll
    for (int cj = 0; cj < 4; ++cj) {
      const bf16_t* vb = &Vt[(16 * cj + l15) * VT_STRIDE + sel * 8];
      o[cj] = wmma_bf16(pf, load_bfrag(vb), o[cj]);
    }
  }

  // --- finalize: divide by row sums, store bf16 in (B*T, H*dh) layout ---
  const float li = 1.f / l_s;
  float lv[8];
#pragma unroll
  for (int v = 0; v < 8; ++v) lv[v] = __shfl(li, v + 8 * sel, 32);
#pragma unroll
  for (int cj = 0; cj < 4; ++cj) {
#pragma unroll
    for (int v = 0; v < 8; ++v) {
      const float val = o[cj][v] * lv[v];
      attn_out[(size_t)(b * T_SEQ + q0 + v + 8 * sel) * C_MODEL + h * D_HEAD +
               16 * cj + l15] = (bf16_t)val;
    }
  }
}

// ---------------------------------------------------------------------------
// Host-side launcher
// ---------------------------------------------------------------------------
extern "C" void kernel_launch(void* const* d_in, const int* in_sizes, int n_in,
                              void* d_out, int out_size, void* d_ws, size_t ws_size,
                              hipStream_t stream) {
  (void)in_sizes; (void)n_in; (void)out_size; (void)ws_size;
  const float* x      = (const float*)d_in[0];
  // d_in[1] is the boolean mask (causality is applied analytically)
  const float* qkv_w  = (const float*)d_in[2];
  const float* qkv_b  = (const float*)d_in[3];
  const float* out_w  = (const float*)d_in[4];
  const float* out_b  = (const float*)d_in[5];
  float* out          = (float*)d_out;

  char* ws = (char*)d_ws;
  const int M = BATCH * T_SEQ;                       // 4096 rows
  bf16_t* x_bf    = (bf16_t*)(ws);                   //  8 MB
  bf16_t* wqkv_bf = (bf16_t*)(ws + (size_t)8  * 1024 * 1024);  // 6 MB
  bf16_t* wout_bf = (bf16_t*)(ws + (size_t)14 * 1024 * 1024);  // 2 MB
  bf16_t* qkv_bf  = (bf16_t*)(ws + (size_t)16 * 1024 * 1024);  // 24 MB
  bf16_t* attn_bf = (bf16_t*)(ws + (size_t)40 * 1024 * 1024);  // 8 MB

  // 1) precision conversion of activations + weights
  {
    int n1 = M * C_MODEL;          // x
    int n2 = QKV_C * C_MODEL;      // qkv_w
    int n3 = C_MODEL * C_MODEL;    // out_w
    cvt_f32_to_bf16<<<(n1 / 4 + 255) / 256, 256, 0, stream>>>(x, x_bf, n1);
    cvt_f32_to_bf16<<<(n2 / 4 + 255) / 256, 256, 0, stream>>>(qkv_w, wqkv_bf, n2);
    cvt_f32_to_bf16<<<(n3 / 4 + 255) / 256, 256, 0, stream>>>(out_w, wout_bf, n3);
  }

  // 2) fused QKV projection: qkv = x @ qkv_w^T + qkv_b   (4096 x 3072)
  {
    const int waves = (M / 64) * (QKV_C / 64);   // 3072
    gemm_wmma<true><<<waves / 8, 256, 0, stream>>>(
        x_bf, wqkv_bf, qkv_b, qkv_bf, nullptr, M, QKV_C, C_MODEL);
  }

  // 3) causal flash attention, all heads
  {
    const int blocks = BATCH * N_HEADS * (T_SEQ / 64);  // 1024
    flash_attn<<<blocks, 128, 0, stream>>>(qkv_bf, attn_bf);
  }

  // 4) output projection: out = attn @ out_w^T + out_b   (4096 x 1024, f32)
  {
    const int waves = (M / 64) * (C_MODEL / 64);  // 1024
    gemm_wmma<false><<<waves / 8, 256, 0, stream>>>(
        attn_bf, wout_bf, out_b, nullptr, out, M, C_MODEL, C_MODEL);
  }
}
